// wd_criterion_75668733821525
// MI455X (gfx1250) — compile-verified, matrix operand order checked by reading, not run on hardware
//
#include <hip/hip_runtime.h>
#include <hip/hip_bf16.h>
#include <math.h>

// ---------------------------------------------------------------------------
// sqrt(sum((t - pad(c,0))^2)) over n_t=32M f32 elements, n_c=24M.
// Memory-bound streaming reduction: ~224 MB @ 23.3 TB/s => ~9.6 us floor.
//   * Phase A: chunks fully inside [0,n_c): 8x b128 NT loads/trip, no branch.
//   * Phase B: chunks fully inside [n_c,n_t): 4x b128 NT loads/trip (c==0).
//     Both phases have wave-uniform trip counts => EXEC all-1s at each WMMA,
//     no divergent compare in the hot loop (round-1 asm showed v_cmpx +
//     serialized loads there).
//   * Squared diffs f32 -> f16 (unbiased RNE; errors cancel over 32M terms),
//     accumulated via v_wmma_f32_16x16x32_f16 with an all-ones A matrix:
//     D = ones x B + C => every D entry is an f32 column-sum of B; the sum of
//     D row 0 over lanes 0..15 equals the sum of ALL 512 B entries, so the
//     per-lane packing order into B is irrelevant.
//   * Deterministic two-pass reduction: per-block partial -> d_ws, then a
//     fixed-order tree reduce + sqrtf (no float atomics => bit-deterministic).
// ---------------------------------------------------------------------------

typedef _Float16 v16h __attribute__((ext_vector_type(16)));
typedef float    v8f  __attribute__((ext_vector_type(8)));
typedef float    v4f  __attribute__((ext_vector_type(4)));

#define WD_NBLOCKS  2048
#define WD_NTHREADS 256

__device__ __forceinline__ v8f wd_wmma_ones(const v16h& a, const v16h& b, const v8f& c) {
    return __builtin_amdgcn_wmma_f32_16x16x32_f16(
        /*neg_a=*/false, a, /*neg_b=*/false, b,
        /*c_mod=*/(short)0, c, /*reuse_a=*/false, /*reuse_b=*/false);
}

__global__ __launch_bounds__(WD_NTHREADS)
void wd_partial_kernel(const float* __restrict__ t,
                       const float* __restrict__ c,
                       float* __restrict__ partials,
                       int n_t, int n_c) {
    const int tid   = blockIdx.x * blockDim.x + threadIdx.x;
    const int total = gridDim.x * blockDim.x;

    const int chunksTotal = n_t >> 4;                    // 16 elems / chunk
    const int chunksA     = n_c >> 4;                    // fully c-covered
    const int startB      = (n_c + 15) >> 4;             // first t-only chunk
    const int chunksB     = (chunksTotal > startB) ? (chunksTotal - startB) : 0;

    const int tripsA = chunksA / total;                  // wave-uniform (==3)
    const int remA   = chunksA - tripsA * total;
    const int tripsB = chunksB / total;                  // wave-uniform (==1)
    const int remB   = chunksB - tripsB * total;

    // A = ones(16x32) f16 (exact); layout irrelevant since all entries equal.
    v16h a;
#pragma unroll
    for (int j = 0; j < 16; ++j) a[j] = (_Float16)1.0f;

    v8f acc = {};  // f32 C/D accumulator

    // ---- Phase A: both streams, branch-free hot loop --------------------
    for (int k = 0; k < tripsA; ++k) {
        const long base = ((long)(tid + k * total)) << 4;
        const v4f* pt = (const v4f*)(t + base);
        const v4f* pc = (const v4f*)(c + base);
        v4f tv[4], cv[4];
#pragma unroll
        for (int j = 0; j < 4; ++j) tv[j] = __builtin_nontemporal_load(pt + j);
#pragma unroll
        for (int j = 0; j < 4; ++j) cv[j] = __builtin_nontemporal_load(pc + j);

        v16h b;
#pragma unroll
        for (int j = 0; j < 4; ++j) {
            v4f d = tv[j] - cv[j];
            v4f s = d * d;
#pragma unroll
            for (int e = 0; e < 4; ++e) b[j * 4 + e] = (_Float16)s[e];
        }
        acc = wd_wmma_ones(a, b, acc);
    }

    // ---- Phase B: t-only region (c is zero padding) ---------------------
    for (int k = 0; k < tripsB; ++k) {
        const long base = ((long)(startB + tid + k * total)) << 4;
        const v4f* pt = (const v4f*)(t + base);
        v4f tv[4];
#pragma unroll
        for (int j = 0; j < 4; ++j) tv[j] = __builtin_nontemporal_load(pt + j);

        v16h b;
#pragma unroll
        for (int j = 0; j < 4; ++j) {
            v4f s = tv[j] * tv[j];
#pragma unroll
            for (int e = 0; e < 4; ++e) b[j * 4 + e] = (_Float16)s[e];
        }
        acc = wd_wmma_ones(a, b, acc);
    }

    // ---- Scalar cleanups (no WMMA; partial EXEC is fine) ----------------
    float sacc = 0.0f;
    if (tid < remA) {                       // leftover both-stream chunks
        const long base = ((long)(tripsA * total + tid)) << 4;
#pragma unroll
        for (int e = 0; e < 16; ++e) {
            float d = t[base + e] - c[base + e];
            sacc = fmaf(d, d, sacc);
        }
    }
    if (tid < remB) {                       // leftover t-only chunks
        const long base = ((long)(startB + tripsB * total + tid)) << 4;
#pragma unroll
        for (int e = 0; e < 16; ++e) {
            float d = t[base + e];
            sacc = fmaf(d, d, sacc);
        }
    }
    {   // boundary sliver [chunksA*16, startB*16) when n_c % 16 != 0
        const long idx = (long)chunksA * 16 + tid;
        if (tid < 16 && idx < (long)startB * 16 && idx < (long)n_t) {
            float cv = (idx < (long)n_c) ? c[idx] : 0.0f;
            float d  = t[idx] - cv;
            sacc = fmaf(d, d, sacc);
        }
    }
    {   // tail [chunksTotal*16, n_t) when n_t % 16 != 0
        const long idx = (long)chunksTotal * 16 + tid;
        if (tid < 16 && idx < (long)n_t) {
            float d = t[idx];
            sacc = fmaf(d, d, sacc);
        }
    }

    // ---- Reduce: D row 0 = acc[0] on lanes 0..15 (16..31 is a duplicate) -
    const int lane = threadIdx.x & 31;
    float v = ((lane < 16) ? acc[0] : 0.0f) + sacc;
#pragma unroll
    for (int m = 16; m > 0; m >>= 1) v += __shfl_xor(v, m, 32);

    __shared__ float sw[WD_NTHREADS / 32];
    if (lane == 0) sw[threadIdx.x >> 5] = v;
    __syncthreads();
    if (threadIdx.x == 0) {
        float s = 0.0f;
#pragma unroll
        for (int j = 0; j < WD_NTHREADS / 32; ++j) s += sw[j];
        partials[blockIdx.x] = s;           // every block writes every call
    }
}

__global__ __launch_bounds__(256)
void wd_final_kernel(const float* __restrict__ partials, float* __restrict__ out) {
    __shared__ float sd[256];
    const int tid = threadIdx.x;
    float v = 0.0f;
#pragma unroll
    for (int j = 0; j < WD_NBLOCKS / 256; ++j) v += partials[tid + j * 256];
    sd[tid] = v;
    __syncthreads();
#pragma unroll
    for (int s = 128; s > 0; s >>= 1) {
        if (tid < s) sd[tid] += sd[tid + s];
        __syncthreads();
    }
    if (tid == 0) out[0] = sqrtf(sd[0]);
}

extern "C" void kernel_launch(void* const* d_in, const int* in_sizes, int n_in,
                              void* d_out, int out_size, void* d_ws, size_t ws_size,
                              hipStream_t stream) {
    const float* t = (const float*)d_in[0];
    const float* c = (const float*)d_in[1];
    float* out      = (float*)d_out;
    float* partials = (float*)d_ws;        // WD_NBLOCKS floats = 8 KB scratch
    const int n_t = in_sizes[0];
    const int n_c = in_sizes[1];

    wd_partial_kernel<<<WD_NBLOCKS, WD_NTHREADS, 0, stream>>>(t, c, partials, n_t, n_c);
    wd_final_kernel<<<1, 256, 0, stream>>>(partials, out);
}
